// MetalAcceleratedAttention_19043884990738
// MI455X (gfx1250) — matrix operand order repaired and checked by reading, not compiled, not run on hardware
//
#include <hip/hip_runtime.h>
#include <stdint.h>

#define B_   2
#define L_   2048
#define D_   1024
#define H_   16
#define DH_  64
#define SCALE_ 0.125f   // DH^-0.5

#define HEAVY_KERNEL_ATTRS \
    __attribute__((amdgpu_flat_work_group_size(256, 256), amdgpu_waves_per_eu(2)))

typedef __attribute__((ext_vector_type(16))) __bf16    v16bf;
typedef __attribute__((ext_vector_type(8)))  float     v8f;
typedef __attribute__((ext_vector_type(4)))  uint32_t  u32x4;

union FragBf {
    v16bf        v;
    unsigned short s[16];
    u32x4        q[2];
};

__device__ __forceinline__ unsigned short f2bf(float f) {
    union { float f; uint32_t u; } c; c.f = f;
    uint32_t u = c.u + 0x7FFFu + ((c.u >> 16) & 1u);   // round-to-nearest-even
    return (unsigned short)(u >> 16);
}

__device__ __forceinline__ u32x4 ld128(const unsigned short* p) {
    return *(const u32x4*)p;
}

__device__ __forceinline__ v8f wmma_bf16(v16bf a, v16bf b, v8f c) {
    // D = A(16x32 bf16) * B(32x16 bf16) + C(16x16 f32)
    return __builtin_amdgcn_wmma_f32_16x16x32_bf16(false, a, false, b, (short)0, c, false, false);
}

// gfx1250 async global->LDS: each active lane copies 16B MEM[ga] -> LDS[lds]
__device__ __forceinline__ void async_ld_b128(uint32_t lds, uint64_t ga) {
    asm volatile("global_load_async_to_lds_b128 %0, %1, off"
                 :: "v"(lds), "v"(ga) : "memory");
}
__device__ __forceinline__ void wait_async0() {
    asm volatile("s_wait_asynccnt 0x0" ::: "memory");
}
// Panel barrier: only drain LDS reads (dscnt); deliberately leave global loads
// (A-fragment prefetch, loadcnt) in flight across the workgroup barrier.
__device__ __forceinline__ void panel_barrier() {
    asm volatile("s_wait_dscnt 0x0\n\t"
                 "s_barrier_signal -1\n\t"
                 "s_barrier_wait -1" ::: "memory");
}

// ---------------- fp32 -> bf16 conversion kernels ----------------

__global__ void cvt_bf16_kernel(const float* __restrict__ src,
                                unsigned short* __restrict__ dst, int n) {
    int i = blockIdx.x * blockDim.x + threadIdx.x;
    int stride = gridDim.x * blockDim.x;
    for (; i < n; i += stride) dst[i] = f2bf(src[i]);
}

// W[K,N] fp32 -> Wt[N,K] bf16 (so B-operand fragments are contiguous in K)
__global__ void cvt_bf16_T_kernel(const float* __restrict__ src,
                                  unsigned short* __restrict__ dst, int K, int N) {
    int i = blockIdx.x * blockDim.x + threadIdx.x;
    int stride = gridDim.x * blockDim.x;
    int total = K * N;
    for (; i < total; i += stride) {
        int k = i / N, n = i - k * N;
        dst[(size_t)n * K + k] = f2bf(src[i]);
    }
}

// ============================================================================
// GEMM core: block (8 waves) computes 256x64; wave w owns rows [m0+32w, +32).
// B panel (64 cols x 32 k = 4KB) staged per block via async->LDS, double-buffered.
// LDS layout: bpanel[buf][col*32 + k]  (ushort), col-major panel.
// ============================================================================

#define GEMM_PROLOGUE(Kd_)                                                        \
    __shared__ __align__(16) unsigned short bpanel[2][64 * 32];                   \
    const int Kd = (Kd_);                                                         \
    int tid  = threadIdx.x;                                                       \
    int wv   = tid >> 5;                                                          \
    int lane = tid & 31;                                                          \
    int half = lane >> 4, ln = lane & 15;                                         \
    uint32_t lds_base = (uint32_t)(uintptr_t)(&bpanel[0][0]);                     \
    /* async staging assignment: thread -> (col, k-offset) */                     \
    int acol = tid >> 2, ako = (tid & 3) * 8;

#define GEMM_ISSUE_B(buf, k0)                                                     \
    async_ld_b128(lds_base + (uint32_t)(buf) * 4096u + (uint32_t)(acol * 64 + ako * 2), \
                  (uint64_t)(uintptr_t)(WT + (size_t)(n0 + acol) * Kd + (k0) + ako))

__device__ __forceinline__ void read_b_frags(const unsigned short* base, int ln, int half,
                                             FragBf* d) {
#pragma unroll
    for (int s = 0; s < 4; ++s) {
        const unsigned short* pp = base + (s * 16 + ln) * 32 + half * 8;
        d[s].q[0] = *(const u32x4*)pp;
        d[s].q[1] = *(const u32x4*)(pp + 16);
    }
}

// ---------------- QKV projection GEMM ----------------
// C[4096,3072] = Xb[4096,1024] @ Wqkv + bias ; scatter Q,K [BH,L,DH], V^T [BH,DH,L]
__global__ void HEAVY_KERNEL_ATTRS
gemm_qkv_kernel(const unsigned short* __restrict__ Xb,
                const unsigned short* __restrict__ WT,   // [3072,1024]
                const float* __restrict__ bias,          // [3072]
                unsigned short* __restrict__ Qb,
                unsigned short* __restrict__ Kb,
                unsigned short* __restrict__ Vt) {
    const int NT = 3072 / 64;
    GEMM_PROLOGUE(1024)
    int nt = blockIdx.x % NT, mblk = blockIdx.x / NT;
    int n0 = nt * 64;
    int m0 = mblk * 256 + wv * 32;

    const unsigned short* a0 = Xb + (size_t)(m0 + ln) * Kd;
    const unsigned short* a1 = Xb + (size_t)(m0 + 16 + ln) * Kd;

    v8f acc[8] = {};
    FragBf a[2], b[4], an[2];
    auto loadA = [&](FragBf* d, int k0) {
        d[0].q[0] = ld128(a0 + k0 + half * 8);
        d[0].q[1] = ld128(a0 + k0 + 16 + half * 8);
        d[1].q[0] = ld128(a1 + k0 + half * 8);
        d[1].q[1] = ld128(a1 + k0 + 16 + half * 8);
    };
    auto domm = [&]() {
#pragma unroll
        for (int mi = 0; mi < 2; ++mi)
#pragma unroll
            for (int s = 0; s < 4; ++s)
                acc[mi * 4 + s] = wmma_bf16(a[mi].v, b[s].v, acc[mi * 4 + s]);
    };

    const int NIT = 1024 / 32;
    GEMM_ISSUE_B(0, 0);
    loadA(a, 0);
    for (int it = 0; it < NIT - 1; ++it) {     // steady state: always prefetch
        wait_async0();                 // own async batch for buf it&1 complete
        panel_barrier();               // dscnt-only drain + workgroup barrier
        GEMM_ISSUE_B((it + 1) & 1, (it + 1) * 32);
        loadA(an, (it + 1) * 32);
        read_b_frags(&bpanel[it & 1][0], ln, half, b);
        domm();
        a[0] = an[0];
        a[1] = an[1];
    }
    wait_async0();                             // peeled final iteration
    panel_barrier();
    read_b_frags(&bpanel[(NIT - 1) & 1][0], ln, half, b);
    domm();

    int three = n0 >> 10;
    int h     = (n0 >> 6) & 15;
#pragma unroll
    for (int s = 0; s < 4; ++s) {
        int dh = s * 16 + ln;
        float bval = bias[n0 + s * 16 + ln];
#pragma unroll
        for (int mi = 0; mi < 2; ++mi)
#pragma unroll
            for (int r = 0; r < 8; ++r) {
                int mrow = m0 + mi * 16 + r + 8 * half;
                int bb = mrow >> 11, l = mrow & (L_ - 1);
                unsigned short v = f2bf(acc[mi * 4 + s][r] + bval);
                size_t bh = (size_t)bb * H_ + h;
                if (three == 0)      Qb[(bh * L_ + l) * DH_ + dh] = v;
                else if (three == 1) Kb[(bh * L_ + l) * DH_ + dh] = v;
                else                 Vt[(bh * DH_ + dh) * L_ + l] = v;
            }
    }
}

// ---------------- output projection GEMM ----------------
__global__ void HEAVY_KERNEL_ATTRS
gemm_proj_kernel(const unsigned short* __restrict__ Ab,   // [4096,1024] bf16
                 const unsigned short* __restrict__ WT,   // [1024,1024] (N,K)
                 const float* __restrict__ bias,          // [1024]
                 float* __restrict__ out) {
    const int N = 1024, NT = N / 64;
    GEMM_PROLOGUE(1024)
    int nt = blockIdx.x % NT, mblk = blockIdx.x / NT;
    int n0 = nt * 64;
    int m0 = mblk * 256 + wv * 32;

    const unsigned short* a0 = Ab + (size_t)(m0 + ln) * Kd;
    const unsigned short* a1 = Ab + (size_t)(m0 + 16 + ln) * Kd;

    v8f acc[8] = {};
    FragBf a[2], b[4], an[2];
    auto loadA = [&](FragBf* d, int k0) {
        d[0].q[0] = ld128(a0 + k0 + half * 8);
        d[0].q[1] = ld128(a0 + k0 + 16 + half * 8);
        d[1].q[0] = ld128(a1 + k0 + half * 8);
        d[1].q[1] = ld128(a1 + k0 + 16 + half * 8);
    };
    auto domm = [&]() {
#pragma unroll
        for (int mi = 0; mi < 2; ++mi)
#pragma unroll
            for (int s = 0; s < 4; ++s)
                acc[mi * 4 + s] = wmma_bf16(a[mi].v, b[s].v, acc[mi * 4 + s]);
    };

    const int NIT = 1024 / 32;
    GEMM_ISSUE_B(0, 0);
    loadA(a, 0);
    for (int it = 0; it < NIT - 1; ++it) {
        wait_async0();
        panel_barrier();
        GEMM_ISSUE_B((it + 1) & 1, (it + 1) * 32);
        loadA(an, (it + 1) * 32);
        read_b_frags(&bpanel[it & 1][0], ln, half, b);
        domm();
        a[0] = an[0];
        a[1] = an[1];
    }
    wait_async0();
    panel_barrier();
    read_b_frags(&bpanel[(NIT - 1) & 1][0], ln, half, b);
    domm();

#pragma unroll
    for (int s = 0; s < 4; ++s) {
        float bval = bias[n0 + s * 16 + ln];
#pragma unroll
        for (int mi = 0; mi < 2; ++mi)
#pragma unroll
            for (int r = 0; r < 8; ++r)
                out[(size_t)(m0 + mi * 16 + r + 8 * half) * N + n0 + s * 16 + ln] =
                    acc[mi * 4 + s][r] + bval;
    }
}

// ---------------- flash attention: one wave per (b,h,32-query block) ----------------
__global__ void HEAVY_KERNEL_ATTRS
attn_kernel(const unsigned short* __restrict__ Qb,
            const unsigned short* __restrict__ Kb,
            const unsigned short* __restrict__ Vt,
            unsigned short* __restrict__ attnb) {
    const int QT = L_ / 32;                    // 64 query blocks per (b,h)
    int wave = (blockIdx.x * blockDim.x + threadIdx.x) >> 5;
    int lane = threadIdx.x & 31;
    int half = lane >> 4, ln = lane & 15;
    int qt = wave % QT, bh = wave / QT;
    if (bh >= B_ * H_) return;                 // wave-uniform
    int q0 = qt * 32;

    const unsigned short* qbase = Qb + (size_t)bh * L_ * DH_;
    const unsigned short* kbase = Kb + (size_t)bh * L_ * DH_;
    const unsigned short* vbase = Vt + (size_t)bh * DH_ * L_;

    // Q^T B-operand fragments for two 16-query tiles (dh chunks 0-31, 32-63)
    FragBf qf[2][2];
#pragma unroll
    for (int u = 0; u < 2; ++u)
#pragma unroll
        for (int c = 0; c < 2; ++c) {
            const unsigned short* p =
                qbase + (size_t)(q0 + u * 16 + ln) * DH_ + c * 32 + half * 8;
            qf[u][c].q[0] = ld128(p);
            qf[u][c].q[1] = ld128(p + 16);
        }

    // K A-operand fragments: [key tile t][dh chunk c] -> kf[2t+c]; double-buffered
    FragBf kf[4], knx[4], vf[4];
    auto loadK = [&](FragBf* d, int kb) {
#pragma unroll
        for (int t = 0; t < 2; ++t)
#pragma unroll
            for (int c = 0; c < 2; ++c) {
                const unsigned short* p = kbase + (size_t)(kb + t * 16 + ln) * DH_ + c * 32;
                d[t * 2 + c].q[0] = ld128(p + half * 8);
                d[t * 2 + c].q[1] = ld128(p + 16 + half * 8);
            }
    };
    auto loadV = [&](FragBf* d, int kb) {
#pragma unroll
        for (int nt = 0; nt < 4; ++nt) {
            const unsigned short* p = vbase + (size_t)(nt * 16 + ln) * L_ + kb + half * 8;
            d[nt].q[0] = ld128(p);
            d[nt].q[1] = ld128(p + 16);
        }
    };

    v8f acc[2][4] = {};
    float m[2]    = {-1e30f, -1e30f};
    float ssum[2] = {0.f, 0.f};

    auto process = [&](int kb) {
#pragma unroll
        for (int u = 0; u < 2; ++u) {
            int qg = q0 + u * 16 + ln;
            // S^T tiles: (16 keys x 16 queries) = K(16x64) @ Q^T(64x16)
            v8f st[2];
#pragma unroll
            for (int t = 0; t < 2; ++t) {
                v8f s = {};
                s = wmma_bf16(kf[t * 2 + 0].v, qf[u][0].v, s);
                s = wmma_bf16(kf[t * 2 + 1].v, qf[u][1].v, s);
                st[t] = s;
            }
            // scale + causal mask + block max
            float pv[16];
            float bmax = -1e30f;
#pragma unroll
            for (int t = 0; t < 2; ++t)
#pragma unroll
                for (int r = 0; r < 8; ++r) {
                    int kidx = kb + t * 16 + r + 8 * half;
                    float sv = st[t][r] * SCALE_;
                    if (kidx > qg) sv = -1e30f;
                    pv[t * 8 + r] = sv;
                    bmax = fmaxf(bmax, sv);
                }
            bmax = fmaxf(bmax, __shfl_xor(bmax, 16, 32));  // merge the two key half-sets
            float mnew = fmaxf(m[u], bmax);
            float scale_old = __expf(m[u] - mnew);
            float ps = 0.f;
            FragBf pf;                                     // P lands in A-fragment layout
#pragma unroll
            for (int e = 0; e < 16; ++e) {
                float pe = __expf(pv[e] - mnew);
                ps += pe;
                pf.s[e] = f2bf(pe);
            }
            ps += __shfl_xor(ps, 16, 32);
            ssum[u] = ssum[u] * scale_old + ps;
            m[u] = mnew;
            // rescale accumulators: out register r belongs to query r + 8*half
#pragma unroll
            for (int r = 0; r < 8; ++r) {
                float f = __shfl(scale_old, r + 8 * half, 32);
                acc[u][0][r] *= f; acc[u][1][r] *= f;
                acc[u][2][r] *= f; acc[u][3][r] *= f;
            }
            // P(16x32) @ V(32x64): 4 WMMAs (V fragments shared across both q tiles)
#pragma unroll
            for (int ntile = 0; ntile < 4; ++ntile)
                acc[u][ntile] = wmma_bf16(pf.v, vf[ntile].v, acc[u][ntile]);
        }
    };

    int nb = (q0 + 63) >> 5;                   // key blocks to cover keys <= q0+31
    loadK(kf, 0);
    for (int blk = 0; blk + 1 < nb; ++blk) {   // steady state: always prefetch next K
        int kb = blk * 32;
        loadV(vf, kb);                         // independent of softmax chain
        loadK(knx, kb + 32);
        process(kb);
#pragma unroll
        for (int i = 0; i < 4; ++i) kf[i] = knx[i];
    }
    {                                          // peeled final block: no prefetch
        int kb = (nb - 1) * 32;
        loadV(vf, kb);
        process(kb);
    }

    int b = bh / H_, h = bh % H_;
    unsigned short* ob = attnb + (size_t)b * L_ * D_ + (size_t)h * DH_;
#pragma unroll
    for (int u = 0; u < 2; ++u) {
        float rden = 1.0f / (ssum[u] + 1e-6f);
#pragma unroll
        for (int r = 0; r < 8; ++r) {
            float f = __shfl(rden, r + 8 * half, 32);
            int qrow = q0 + u * 16 + r + 8 * half;
#pragma unroll
            for (int ntile = 0; ntile < 4; ++ntile)
                ob[(size_t)qrow * D_ + ntile * 16 + ln] = f2bf(acc[u][ntile][r] * f);
        }
    }
}

// ---------------- host-side launch ----------------

extern "C" void kernel_launch(void* const* d_in, const int* in_sizes, int n_in,
                              void* d_out, int out_size, void* d_ws, size_t ws_size,
                              hipStream_t stream) {
    const float* x      = (const float*)d_in[0];   // [B,L,D]
    const float* W_qkv  = (const float*)d_in[1];   // [D,3D]
    const float* b_qkv  = (const float*)d_in[2];   // [3D]
    const float* W_proj = (const float*)d_in[3];   // [D,D]
    const float* b_proj = (const float*)d_in[4];   // [D]
    float* out = (float*)d_out;                    // [B,L,D] fp32

    // carve workspace (bf16 buffers), 256B aligned
    uintptr_t p = ((uintptr_t)d_ws + 255) & ~(uintptr_t)255;
    auto take = [&](size_t elems) {
        unsigned short* r = (unsigned short*)p;
        p = (p + elems * sizeof(unsigned short) + 255) & ~(uintptr_t)255;
        return r;
    };
    const size_t M  = (size_t)B_ * L_;             // 4096
    unsigned short* xb     = take(M * D_);         // x bf16 (reused as attn out)
    unsigned short* wqkvT  = take((size_t)3 * D_ * D_);
    unsigned short* wprojT = take((size_t)D_ * D_);
    unsigned short* Qb     = take((size_t)B_ * H_ * L_ * DH_);
    unsigned short* Kb     = take((size_t)B_ * H_ * L_ * DH_);
    unsigned short* Vt     = take((size_t)B_ * H_ * L_ * DH_);
    unsigned short* attnb  = xb;                   // alias: x dead after QKV GEMM

    // 1) conversions
    cvt_bf16_kernel<<<2048, 256, 0, stream>>>(x, xb, (int)(M * D_));
    cvt_bf16_T_kernel<<<3072, 256, 0, stream>>>(W_qkv,  wqkvT,  D_, 3 * D_);
    cvt_bf16_T_kernel<<<1024, 256, 0, stream>>>(W_proj, wprojT, D_, D_);

    // 2) QKV GEMM: 16 m-blocks(256 rows) * 48 n-strips = 768 blocks of 8 waves
    gemm_qkv_kernel<<<768, 256, 0, stream>>>(xb, wqkvT, b_qkv, Qb, Kb, Vt);

    // 3) flash attention: B*H*(L/32) = 2048 waves / 8 per block
    attn_kernel<<<256, 256, 0, stream>>>(Qb, Kb, Vt, attnb);

    // 4) output projection: 16 m-blocks * 16 n-strips = 256 blocks
    gemm_proj_kernel<<<256, 256, 0, stream>>>(attnb, wprojT, b_proj, out);
}
